// DEQEQFusionBlock_80642305949812
// MI455X (gfx1250) — matrix operand order repaired and checked by reading, not compiled, not run on hardware
//
#include <hip/hip_runtime.h>
#include <hip/hip_bf16.h>
#include <cstdint>

// ---------------- problem constants ----------------
#define C_CH   256
#define T_LEN  512
#define TPAD   (T_LEN + 2)         // one zero column on each side for SAME conv
#define NB_    4
#define CH3    768
#define D_FLAT (CH3 * T_LEN)       // 393216
#define M_AND_ 6
#define F_THRES_ 30
#define LAM_   1e-4f
#define BETA_  1.0f
#define GRAMNB 32
#define NPAIR  21

typedef _Float16 v8h  __attribute__((ext_vector_type(8)));
typedef _Float16 v16h __attribute__((ext_vector_type(16)));
typedef float    v8f  __attribute__((ext_vector_type(8)));

__device__ __forceinline__ v8h ld8(const _Float16* p) { return *(const v8h*)p; }
__device__ __forceinline__ v16h join16(v8h lo, v8h hi) {
    union { v16h v; v8h h[2]; } u;
    u.h[0] = lo; u.h[1] = hi;
    return u.v;
}
__device__ __forceinline__ float sigm(float x) { return 1.f / (1.f + __expf(-x)); }

// ---------------- zero-fill (capture-safe init) ----------------
__global__ void k_zero(float* __restrict__ p, size_t n) {
    size_t i = (size_t)blockIdx.x * blockDim.x + threadIdx.x;
    size_t stride = (size_t)gridDim.x * blockDim.x;
    for (; i < n; i += stride) p[i] = 0.f;
}

// ---------------- weight convert: (Co,Ci,3) f32 -> (3,Co,Ci) f16 ----------------
__global__ void k_wcvt(const float* __restrict__ W, _Float16* __restrict__ Wt) {
    int tid = blockIdx.x * blockDim.x + threadIdx.x;   // 3*256*256 = 196608
    if (tid >= 3 * C_CH * C_CH) return;
    int ci  = tid & 255;
    int co  = (tid >> 8) & 255;
    int tap = tid >> 16;
    Wt[tid] = (_Float16)W[((size_t)co * C_CH + ci) * 3 + tap];
}

// ---------------- slice z (B,768,T) f32 -> xT (B,TPAD,256) f16 (interior cols) ----------------
__global__ void k_sliceT(const float* __restrict__ z, int chanOff, _Float16* __restrict__ xT) {
    int tid = blockIdx.x * blockDim.x + threadIdx.x;   // 4*512*256 = 524288
    if (tid >= NB_ * T_LEN * C_CH) return;
    int c = tid & 255;
    int t = (tid >> 8) & 511;
    int b = tid >> 17;
    xT[((size_t)b * TPAD + t + 1) * C_CH + c] =
        (_Float16)z[((size_t)b * CH3 + chanOff + c) * T_LEN + t];
}

// ---------------- WMMA conv1d (K=3, SAME via padded input) ----------------
// MODE 0: out = tanh(conv(x,W0)) * sigmoid(conv(x,W1)) + addA      (gated residual pre-cln)
// MODE 1: out = sigmoid(conv(x,W0)) * conv(x,W1)                   (fused store)
// MODE 2: out += sigmoid(conv(x,W0)) * conv(x,W1)                  (fused accumulate)
// MODE 3: out = conv(x,W0) + addA + addB                           (self path pre-cln)
template <int MODE>
__global__ __launch_bounds__(256)
void k_conv(const _Float16* __restrict__ xT,
            const _Float16* __restrict__ W0,
            const _Float16* __restrict__ W1,
            const float* __restrict__ addA,
            const float* __restrict__ addB,
            float* __restrict__ out) {
    const int wid  = (blockIdx.x << 3) | (threadIdx.x >> 5);  // 2048 waves total
    const int lane = threadIdx.x & 31;
    const int b  = wid >> 9;                                  // 512 tiles / batch
    const int mt = (wid >> 5) & 15;                           // 16 co-tiles
    const int nt = wid & 31;                                  // 32 t-tiles
    const int m0 = mt << 4, n0 = nt << 4;
    const int lm = lane & 15, lh = lane >> 4;

    v8f acc0, acc1;
    for (int i = 0; i < 8; ++i) { acc0[i] = 0.f; acc1[i] = 0.f; }

    const _Float16* wrow0 = W0 + (size_t)(m0 + lm) * C_CH;
    const _Float16* wrow1 = (MODE == 3) ? nullptr : (W1 + (size_t)(m0 + lm) * C_CH);

    for (int tap = 0; tap < 3; ++tap) {
        // padded column index: (t + 1) = n0 + lm + tap - 1 + 1; always in [0, TPAD)
        const _Float16* xcol = xT + ((size_t)b * TPAD + (n0 + lm + tap)) * C_CH;
        const _Float16* w0t = wrow0 + (size_t)tap * (C_CH * C_CH);
        const _Float16* w1t = (MODE == 3) ? nullptr : (wrow1 + (size_t)tap * (C_CH * C_CH));
#pragma unroll
        for (int kb = 0; kb < 8; ++kb) {
            const int k0 = kb << 5;
            // B fragment: column t of x, K = k0 + 8*lh + [0..7] and k0 + 16 + 8*lh + [0..7] (2x b128)
            v16h bf = join16(ld8(xcol + k0 + 8 * lh), ld8(xcol + k0 + 16 + 8 * lh));
            // A fragment: row co of W, K = k0 + 8*lh + [0..7] and k0 + 16 + 8*lh + [0..7]
            v16h a0 = join16(ld8(w0t + k0 + 8 * lh), ld8(w0t + k0 + 16 + 8 * lh));
            acc0 = __builtin_amdgcn_wmma_f32_16x16x32_f16(
                false, a0, false, bf, (short)0, acc0, false, false);
            if (MODE != 3) {
                v16h a1 = join16(ld8(w1t + k0 + 8 * lh), ld8(w1t + k0 + 16 + 8 * lh));
                acc1 = __builtin_amdgcn_wmma_f32_16x16x32_f16(
                    false, a1, false, bf, (short)0, acc1, false, false);
            }
        }
    }

    const int t = n0 + lm;
#pragma unroll
    for (int r = 0; r < 8; ++r) {
        const int co = m0 + r + 8 * lh;
        const size_t idx = ((size_t)b * C_CH + co) * T_LEN + t;
        float v;
        if (MODE == 0)      v = tanhf(acc0[r]) * sigm(acc1[r]) + addA[idx];
        else if (MODE == 1) v = sigm(acc0[r]) * acc1[r];
        else if (MODE == 2) v = out[idx] + sigm(acc0[r]) * acc1[r];
        else                v = acc0[r] + addA[idx] + addB[idx];
        out[idx] = v;
    }
}

// ---------------- channel layernorm: mean/var over C per (b,t) column ----------------
// writes f32 into fout slice [b, chanOff + c, t] and optionally f16 (B,TPAD,C) transposed
__global__ __launch_bounds__(256)
void k_cln(const float* __restrict__ pre,
           const float* __restrict__ gamma,
           const float* __restrict__ beta,
           float* __restrict__ fout, int chanOff,
           _Float16* __restrict__ outT) {
    const int wid  = (blockIdx.x << 3) | (threadIdx.x >> 5);  // 2048 waves = B*T
    const int lane = threadIdx.x & 31;
    const int b = wid >> 9;
    const int t = wid & 511;
    float v[8];
    float s = 0.f, ss = 0.f;
#pragma unroll
    for (int j = 0; j < 8; ++j) {
        const int c = lane + 32 * j;
        const float x = pre[((size_t)b * C_CH + c) * T_LEN + t];
        v[j] = x; s += x; ss += x * x;
    }
#pragma unroll
    for (int off = 16; off > 0; off >>= 1) {
        s  += __shfl_xor(s,  off, 32);
        ss += __shfl_xor(ss, off, 32);
    }
    const float mu  = s * (1.f / C_CH);
    const float var = ss * (1.f / C_CH) - mu * mu;
    const float rs  = rsqrtf(var + 1e-5f);
#pragma unroll
    for (int j = 0; j < 8; ++j) {
        const int c = lane + 32 * j;
        const float y = gamma[c] * (v[j] - mu) * rs + beta[c];
        fout[((size_t)b * CH3 + chanOff + c) * T_LEN + t] = y;
        if (outT) outT[((size_t)b * TPAD + t + 1) * C_CH + c] = (_Float16)y;
    }
}

// ---------------- Anderson helpers ----------------
__global__ void k_toslot(const float* __restrict__ src, float* __restrict__ dst, int slot) {
    const size_t tid = (size_t)blockIdx.x * blockDim.x + threadIdx.x;
    if (tid >= (size_t)NB_ * D_FLAT) return;
    const int b = (int)(tid / D_FLAT);
    const int d = (int)(tid % D_FLAT);
    dst[((size_t)b * M_AND_ + slot) * D_FLAT + d] = src[tid];
}

__global__ void k_fromslot(const float* __restrict__ src, int slot, float* __restrict__ dst) {
    const size_t tid = (size_t)blockIdx.x * blockDim.x + threadIdx.x;
    if (tid >= (size_t)NB_ * D_FLAT) return;
    const int b = (int)(tid / D_FLAT);
    const int d = (int)(tid % D_FLAT);
    dst[tid] = src[((size_t)b * M_AND_ + slot) * D_FLAT + d];
}

// Stage 1: fixed-order partial dot products for the upper-triangular Gram matrix.
__global__ __launch_bounds__(256)
void k_gram(const float* __restrict__ X, const float* __restrict__ F,
            int n, float* __restrict__ partials) {
    const int gb  = blockIdx.x;                 // NB_*NPAIR*GRAMNB blocks
    const int blk = gb & (GRAMNB - 1);
    const int p   = (gb >> 5) % NPAIR;
    const int b   = gb / (GRAMNB * NPAIR);
    int i = 0, jj = p;
    while (jj >= M_AND_ - i) { jj -= M_AND_ - i; ++i; }
    const int j = i + jj;
    __shared__ float red[256];
    float s = 0.f;
    if (i < n && j < n) {
        const float* Fi = F + ((size_t)b * M_AND_ + i) * D_FLAT;
        const float* Xi = X + ((size_t)b * M_AND_ + i) * D_FLAT;
        const float* Fj = F + ((size_t)b * M_AND_ + j) * D_FLAT;
        const float* Xj = X + ((size_t)b * M_AND_ + j) * D_FLAT;
        for (int d = blk * 256 + threadIdx.x; d < D_FLAT; d += GRAMNB * 256)
            s += (Fi[d] - Xi[d]) * (Fj[d] - Xj[d]);
    }
    red[threadIdx.x] = s;
    __syncthreads();
    for (int st = 128; st > 0; st >>= 1) {
        if ((int)threadIdx.x < st) red[threadIdx.x] += red[threadIdx.x + st];
        __syncthreads();
    }
    if (threadIdx.x == 0) partials[((size_t)(b * NPAIR) + p) * GRAMNB + blk] = red[0];
}

// Stage 2: finish Gram sums, build 7x7 bordered system, Gauss-Jordan w/ partial pivot.
__global__ void k_solve(const float* __restrict__ partials, int n, float* __restrict__ alpha) {
    const int b = blockIdx.x;
    float gram[M_AND_][M_AND_];
    int p = 0;
    for (int i = 0; i < M_AND_; ++i)
        for (int j = i; j < M_AND_; ++j) {
            float s = 0.f;
            const float* pp = partials + ((size_t)(b * NPAIR) + p) * GRAMNB;
            for (int q = 0; q < GRAMNB; ++q) s += pp[q];
            gram[i][j] = s; gram[j][i] = s;
            ++p;
        }
    float A[7][8];
    for (int r = 0; r < 7; ++r)
        for (int c = 0; c < 8; ++c) A[r][c] = 0.f;
    for (int i = 0; i < M_AND_; ++i) {
        const float mi = (i < n) ? 1.f : 0.f;
        A[0][1 + i] = mi; A[1 + i][0] = mi;
        for (int j = 0; j < M_AND_; ++j) {
            const float mj = (j < n) ? 1.f : 0.f;
            float h = gram[i][j] * mi * mj;
            if (i == j) h += (mi > 0.f) ? LAM_ : 1.f;
            A[1 + i][1 + j] = h;
        }
    }
    A[0][7] = 1.f;  // y = e0
    for (int r = 0; r < 7; ++r) {
        int piv = r; float best = fabsf(A[r][r]);
        for (int q = r + 1; q < 7; ++q) {
            const float vv = fabsf(A[q][r]);
            if (vv > best) { best = vv; piv = q; }
        }
        if (piv != r)
            for (int c = 0; c < 8; ++c) { const float tmp = A[r][c]; A[r][c] = A[piv][c]; A[piv][c] = tmp; }
        const float inv = 1.f / A[r][r];
        for (int c = 0; c < 8; ++c) A[r][c] *= inv;
        for (int q = 0; q < 7; ++q) {
            if (q == r) continue;
            const float f = A[q][r];
            if (f != 0.f)
                for (int c = 0; c < 8; ++c) A[q][c] -= f * A[r][c];
        }
    }
    for (int j = 0; j < M_AND_; ++j) alpha[b * M_AND_ + j] = A[1 + j][7];
}

// newx = beta*sum(a_j F_j) + (1-beta)*sum(a_j X_j); write to zbuf and X[idx]
__global__ __launch_bounds__(256)
void k_mix(float* __restrict__ X, const float* __restrict__ F,
           const float* __restrict__ alpha, int idx, float* __restrict__ zbuf) {
    const size_t tid = (size_t)blockIdx.x * blockDim.x + threadIdx.x;
    if (tid >= (size_t)NB_ * D_FLAT) return;
    const int b = (int)(tid / D_FLAT);
    const int d = (int)(tid % D_FLAT);
    float sF = 0.f, sX = 0.f;
#pragma unroll
    for (int j = 0; j < M_AND_; ++j) {
        const float a = alpha[b * M_AND_ + j];
        sF += a * F[((size_t)b * M_AND_ + j) * D_FLAT + d];
        sX += a * X[((size_t)b * M_AND_ + j) * D_FLAT + d];
    }
    const float nx = BETA_ * sF + (1.f - BETA_) * sX;
    zbuf[tid] = nx;
    X[((size_t)b * M_AND_ + idx) * D_FLAT + d] = nx;
}

// ---------------- host-side orchestration ----------------
struct Ctx {
    const float *feat0, *feat1, *feat2;
    const float *g0, *b0, *g1, *b1, *gf, *bf;
    _Float16 *Wf0, *Wg0, *Wf1, *Wg1, *Wself, *Wp0, *Wgt0, *Wp1, *Wgt1;
    _Float16 *xT0, *xT1, *xTf, *o0T, *o1T;
    float *pre, *fused, *zbuf, *fbuf, *X, *F, *partials, *alpha;
};

static void eval_f(const Ctx& c, const float* zin, float* fout, hipStream_t s) {
    k_sliceT<<<2048, 256, 0, s>>>(zin, 0,   c.xT0);
    k_sliceT<<<2048, 256, 0, s>>>(zin, 256, c.xT1);
    k_sliceT<<<2048, 256, 0, s>>>(zin, 512, c.xTf);
    k_conv<0><<<256, 256, 0, s>>>(c.xT0, c.Wf0, c.Wg0, c.feat0, (const float*)nullptr, c.pre);
    k_cln<<<256, 256, 0, s>>>(c.pre, c.g0, c.b0, fout, 0, c.o0T);
    k_conv<0><<<256, 256, 0, s>>>(c.xT1, c.Wf1, c.Wg1, c.feat1, (const float*)nullptr, c.pre);
    k_cln<<<256, 256, 0, s>>>(c.pre, c.g1, c.b1, fout, 256, c.o1T);
    k_conv<1><<<256, 256, 0, s>>>(c.o0T, c.Wgt0, c.Wp0, (const float*)nullptr, (const float*)nullptr, c.fused);
    k_conv<2><<<256, 256, 0, s>>>(c.o1T, c.Wgt1, c.Wp1, (const float*)nullptr, (const float*)nullptr, c.fused);
    k_conv<3><<<256, 256, 0, s>>>(c.xTf, c.Wself, (const _Float16*)nullptr, c.fused, c.feat2, c.pre);
    k_cln<<<256, 256, 0, s>>>(c.pre, c.gf, c.bf, fout, 512, (_Float16*)nullptr);
}

extern "C" void kernel_launch(void* const* d_in, const int* in_sizes, int n_in,
                              void* d_out, int out_size, void* d_ws, size_t ws_size,
                              hipStream_t stream) {
    (void)in_sizes; (void)n_in; (void)out_size; (void)ws_size;
    Ctx c;
    c.feat0 = (const float*)d_in[0];
    c.feat1 = (const float*)d_in[1];
    c.feat2 = (const float*)d_in[2];
    const float* Wf0s  = (const float*)d_in[3];
    const float* Wg0s  = (const float*)d_in[4];
    c.g0 = (const float*)d_in[5];  c.b0 = (const float*)d_in[6];
    const float* Wf1s  = (const float*)d_in[7];
    const float* Wg1s  = (const float*)d_in[8];
    c.g1 = (const float*)d_in[9];  c.b1 = (const float*)d_in[10];
    const float* Wss   = (const float*)d_in[11];
    const float* Wp0s  = (const float*)d_in[12];
    const float* Wgt0s = (const float*)d_in[13];
    const float* Wp1s  = (const float*)d_in[14];
    const float* Wgt1s = (const float*)d_in[15];
    c.gf = (const float*)d_in[16]; c.bf = (const float*)d_in[17];

    size_t off = 0;
    auto take = [&](size_t bytes) -> char* {
        off = (off + 255) & ~(size_t)255;
        char* p = (char*)d_ws + off;
        off += bytes;
        return p;
    };
    const size_t WTB  = (size_t)3 * C_CH * C_CH * sizeof(_Float16);
    c.Wf0   = (_Float16*)take(WTB); c.Wg0  = (_Float16*)take(WTB);
    c.Wf1   = (_Float16*)take(WTB); c.Wg1  = (_Float16*)take(WTB);
    c.Wself = (_Float16*)take(WTB);
    c.Wp0   = (_Float16*)take(WTB); c.Wgt0 = (_Float16*)take(WTB);
    c.Wp1   = (_Float16*)take(WTB); c.Wgt1 = (_Float16*)take(WTB);
    // NOTE: XTB is a multiple of 256 bytes, so the five xT buffers are contiguous.
    const size_t XTB = (size_t)NB_ * TPAD * C_CH * sizeof(_Float16);
    c.xT0 = (_Float16*)take(XTB); c.xT1 = (_Float16*)take(XTB); c.xTf = (_Float16*)take(XTB);
    c.o0T = (_Float16*)take(XTB); c.o1T = (_Float16*)take(XTB);
    const size_t PREB = (size_t)NB_ * C_CH * T_LEN * sizeof(float);
    c.pre   = (float*)take(PREB);
    c.fused = (float*)take(PREB);
    const size_t ZB = (size_t)NB_ * D_FLAT * sizeof(float);
    c.zbuf = (float*)take(ZB);
    c.fbuf = (float*)take(ZB);
    const size_t HB = (size_t)NB_ * M_AND_ * D_FLAT * sizeof(float);
    c.X = (float*)take(HB);
    c.F = (float*)take(HB);
    c.partials = (float*)take((size_t)NB_ * NPAIR * GRAMNB * sizeof(float));
    c.alpha    = (float*)take((size_t)NB_ * M_AND_ * sizeof(float));

    // weight conversion (f32 -> f16, tap-major)
    k_wcvt<<<768, 256, 0, stream>>>(Wf0s,  c.Wf0);
    k_wcvt<<<768, 256, 0, stream>>>(Wg0s,  c.Wg0);
    k_wcvt<<<768, 256, 0, stream>>>(Wf1s,  c.Wf1);
    k_wcvt<<<768, 256, 0, stream>>>(Wg1s,  c.Wg1);
    k_wcvt<<<768, 256, 0, stream>>>(Wss,   c.Wself);
    k_wcvt<<<768, 256, 0, stream>>>(Wp0s,  c.Wp0);
    k_wcvt<<<768, 256, 0, stream>>>(Wgt0s, c.Wgt0);
    k_wcvt<<<768, 256, 0, stream>>>(Wp1s,  c.Wp1);
    k_wcvt<<<768, 256, 0, stream>>>(Wgt1s, c.Wgt1);

    // zero-init: z buffer, X/F history, and the 5 padded f16 activation buffers
    // (zeroing the xT buffers once establishes the永-zero pad columns; slice/cln
    // kernels only ever write interior columns afterwards)
    k_zero<<<4096, 256, 0, stream>>>(c.zbuf, (size_t)NB_ * D_FLAT);
    k_zero<<<4096, 256, 0, stream>>>(c.X, (size_t)NB_ * M_AND_ * D_FLAT);
    k_zero<<<4096, 256, 0, stream>>>(c.F, (size_t)NB_ * M_AND_ * D_FLAT);
    k_zero<<<4096, 256, 0, stream>>>((float*)c.xT0, (5 * XTB) / sizeof(float));

    const int NTH = 6144;  // (NB_*D_FLAT)/256 blocks

    // warm-up: f0 = f(0); f1 = f(f0)
    eval_f(c, c.zbuf, c.fbuf, stream);                       // fbuf = f0
    k_toslot<<<NTH, 256, 0, stream>>>(c.fbuf, c.F, 0);       // F[:,0] = f0 (X[:,0] = 0)
    eval_f(c, c.fbuf, c.zbuf, stream);                       // zbuf = f1
    k_toslot<<<NTH, 256, 0, stream>>>(c.fbuf, c.X, 1);       // X[:,1] = f0
    k_toslot<<<NTH, 256, 0, stream>>>(c.zbuf, c.F, 1);       // F[:,1] = f1

    // Anderson iterations k = 2 .. threshold-1
    for (int k = 2; k < F_THRES_; ++k) {
        const int n   = (k < M_AND_) ? k : M_AND_;
        const int idx = k % M_AND_;
        k_gram<<<NB_ * NPAIR * GRAMNB, 256, 0, stream>>>(c.X, c.F, n, c.partials);
        k_solve<<<NB_, 1, 0, stream>>>(c.partials, n, c.alpha);
        k_mix<<<NTH, 256, 0, stream>>>(c.X, c.F, c.alpha, idx, c.zbuf);
        eval_f(c, c.zbuf, c.fbuf, stream);
        k_toslot<<<NTH, 256, 0, stream>>>(c.fbuf, c.F, idx);
    }

    // z_star = F[:, (threshold-1) % m]; output = f(z_star)
    k_fromslot<<<NTH, 256, 0, stream>>>(c.F, (F_THRES_ - 1) % M_AND_, c.zbuf);
    eval_f(c, c.zbuf, (float*)d_out, stream);
}
